// DistributedGraphSAGE_82703890251957
// MI455X (gfx1250) — compile-verified, matrix operand order checked by reading, not run on hardware
//
#include <hip/hip_runtime.h>
#include <hip/hip_bf16.h>

// ---------------------------------------------------------------------------
// GraphSAGE (3-layer) for MI455X / gfx1250.
//  - GEMMs via V_WMMA_F32_16X16X32_BF16 (wave32, 16x16 tiles, f32 accum)
//  - f32 -> bf16 via native casts (v_cvt_pk_bf16_f32) instead of bit-twiddles
//  - edge aggregation via global_atomic_add_f32 (all buffers fit in 192MB L2)
// ---------------------------------------------------------------------------

typedef __attribute__((ext_vector_type(16))) __bf16 v16bf;
typedef __attribute__((ext_vector_type(8)))  float  v8f;

#define NNODES 50000
#define NEDGES 800000
#define DIN    128
#define HDIM   128
#define DOUTL  64
#define NGRAPH 64

// ---------------------------------------------------------------------------
// Pack weight matrix W [dout x din] (row-major, f32) into per-lane bf16
// B-fragments for V_WMMA_F32_16X16X32_BF16.
// Fragment f = kt*NT + nt holds B[k][n] for k in [kt*32, kt*32+32),
// n in [nt*16, nt*16+16). Lane l: n = nt*16 + (l&15), g = l>>4,
// element e (0..15): K = kt*32 + (e<8 ? e : e+8) + 8*g.
// Fragment = 32 lanes * 16 bf16 = 512 elements (1 KB); lane stride 32 B.
// ---------------------------------------------------------------------------
__global__ void pack_weights_kernel(const float* __restrict__ W,
                                    __bf16* __restrict__ P,
                                    int NT, int KT, int din) {
    int t = blockIdx.x * blockDim.x + threadIdx.x;
    int total = NT * KT * 32;
    if (t >= total) return;
    int lane = t & 31;
    int frag = t >> 5;
    int kt = frag / NT;
    int nt = frag % NT;
    int g  = lane >> 4;
    int n  = nt * 16 + (lane & 15);
    __bf16* dst = P + (size_t)frag * 512 + (size_t)lane * 16;
    #pragma unroll
    for (int e = 0; e < 16; ++e) {
        int K = kt * 32 + ((e < 8) ? e : (e + 8)) + 8 * g;
        dst[e] = (__bf16)W[(size_t)n * din + K];
    }
}

// ---------------------------------------------------------------------------
// In-degree (float) of every node.
// ---------------------------------------------------------------------------
__global__ void degree_kernel(const int* __restrict__ dst, float* __restrict__ cnt, int E) {
    int stride = gridDim.x * blockDim.x;
    for (int t = blockIdx.x * blockDim.x + threadIdx.x; t < E; t += stride)
        atomicAdd(cnt + dst[t], 1.0f);
}

// ---------------------------------------------------------------------------
// Edge scatter-add: aggr[dst] += x[src], d = 128 (fixed for all 3 layers).
// 32 lanes per edge, float4 gather per lane, 4 f32 atomics.
// ---------------------------------------------------------------------------
__global__ void scatter_add_kernel(const float* __restrict__ x,
                                   const int* __restrict__ src,
                                   const int* __restrict__ dst,
                                   float* __restrict__ aggr, int E) {
    int stride = gridDim.x * blockDim.x;
    int total  = E * 32;
    for (int t = blockIdx.x * blockDim.x + threadIdx.x; t < total; t += stride) {
        int e    = t >> 5;
        int lane = t & 31;
        int s = src[e], d = dst[e];
        const float4 v = *(const float4*)(x + (size_t)s * 128 + lane * 4);
        float* ap = aggr + (size_t)d * 128 + lane * 4;
        atomicAdd(ap + 0, v.x);
        atomicAdd(ap + 1, v.y);
        atomicAdd(ap + 2, v.z);
        atomicAdd(ap + 3, v.w);
    }
}

// ---------------------------------------------------------------------------
// Fused SAGE layer: out = norm( (aggr/deg) @ Wl^T + b + xin @ Wr^T ), opt ReLU.
// One wave (32 threads) per 16-row block; din=128 (KT=4), dout=NT*16.
// C layout (16x16 f32): VGPR r, lanes 0-15 -> row r, lanes 16-31 -> row r+8,
// column = lane & 15.
// ---------------------------------------------------------------------------
template<int NT, bool RELU>
__global__ __launch_bounds__(32)
void sage_gemm_norm_kernel(const float* __restrict__ xin,    // [N,128]
                           const float* __restrict__ aggr,   // [N,128] sums
                           const float* __restrict__ cnt,    // [N]
                           const __bf16* __restrict__ Pl,    // packed Wl
                           const __bf16* __restrict__ Pr,    // packed Wr
                           const float* __restrict__ bias,   // [NT*16]
                           float* __restrict__ out) {        // [N, NT*16]
    const int lane = threadIdx.x;
    const int base = blockIdx.x * 16;       // N % 16 == 0, no bounds check
    const int col  = lane & 15;
    const int half = lane >> 4;
    const int arow = base + col;            // row this lane's A-fragment feeds

    const float inv = 1.0f / fmaxf(cnt[arow], 1.0f);

    v8f c[NT];
    #pragma unroll
    for (int i = 0; i < NT; ++i) c[i] = (v8f){};

    #pragma unroll
    for (int kt = 0; kt < 4; ++kt) {
        // ---- A fragments: two contiguous 8-float runs per lane, native cvt --
        const int k0 = kt * 32 + 8 * half;
        union { v16bf v; __bf16 b[16]; } aA, aX;
        const float* pa = aggr + (size_t)arow * 128 + k0;
        const float* px = xin  + (size_t)arow * 128 + k0;
        #pragma unroll
        for (int e = 0; e < 8; ++e) {
            aA.b[e]     = (__bf16)(pa[e]      * inv);
            aA.b[e + 8] = (__bf16)(pa[e + 16] * inv);
            aX.b[e]     = (__bf16)px[e];
            aX.b[e + 8] = (__bf16)px[e + 16];
        }
        // ---- B fragments + WMMA -------------------------------------------
        #pragma unroll
        for (int nt = 0; nt < NT; ++nt) {
            const size_t fo = (size_t)(kt * NT + nt) * 512;
            union { v16bf v; uint4 q[2]; } bl, br;
            const uint4* bpl = (const uint4*)(Pl + fo) + lane * 2;
            const uint4* bpr = (const uint4*)(Pr + fo) + lane * 2;
            bl.q[0] = bpl[0]; bl.q[1] = bpl[1];
            br.q[0] = bpr[0]; br.q[1] = bpr[1];
            c[nt] = __builtin_amdgcn_wmma_f32_16x16x32_bf16(
                        false, aA.v, false, bl.v, (short)0, c[nt], false, false);
            c[nt] = __builtin_amdgcn_wmma_f32_16x16x32_bf16(
                        false, aX.v, false, br.v, (short)0, c[nt], false, false);
        }
    }

    // ---- bias + row sum-of-squares ----------------------------------------
    float ss[8];
    #pragma unroll
    for (int r = 0; r < 8; ++r) ss[r] = 0.0f;
    #pragma unroll
    for (int nt = 0; nt < NT; ++nt) {
        const float b = bias[nt * 16 + col];
        #pragma unroll
        for (int r = 0; r < 8; ++r) {
            float v = c[nt][r] + b;
            c[nt][r] = v;
            ss[r] += v * v;
        }
    }
    // reduce over the 16 lanes sharing each row (masks stay within a half)
    #pragma unroll
    for (int m = 1; m < 16; m <<= 1) {
        #pragma unroll
        for (int r = 0; r < 8; ++r) ss[r] += __shfl_xor(ss[r], m, 32);
    }
    float sc[8];
    #pragma unroll
    for (int r = 0; r < 8; ++r) sc[r] = 1.0f / fmaxf(sqrtf(ss[r]), 1e-12f);

    // ---- normalize, ReLU, store -------------------------------------------
    #pragma unroll
    for (int nt = 0; nt < NT; ++nt) {
        #pragma unroll
        for (int r = 0; r < 8; ++r) {
            float v = c[nt][r] * sc[r];
            if (RELU) v = fmaxf(v, 0.0f);
            out[(size_t)(base + r + 8 * half) * (NT * 16) + nt * 16 + col] = v;
        }
    }
}

// ---------------------------------------------------------------------------
// Global mean pool: accumulate h3 [N,64] into pool [G,64] and node counts.
// ---------------------------------------------------------------------------
__global__ void pool_accum_kernel(const float* __restrict__ h,
                                  const int* __restrict__ batch,
                                  float* __restrict__ pool,
                                  float* __restrict__ gcnt, int N) {
    int stride = gridDim.x * blockDim.x;
    int total  = N * 16;
    for (int t = blockIdx.x * blockDim.x + threadIdx.x; t < total; t += stride) {
        int n  = t >> 4;
        int c4 = t & 15;
        const float4 v = *(const float4*)(h + (size_t)n * 64 + c4 * 4);
        int b = batch[n];
        float* pp = pool + (size_t)b * 64 + c4 * 4;
        atomicAdd(pp + 0, v.x);
        atomicAdd(pp + 1, v.y);
        atomicAdd(pp + 2, v.z);
        atomicAdd(pp + 3, v.w);
        if (c4 == 0) atomicAdd(gcnt + b, 1.0f);
    }
}

__global__ void pool_final_kernel(const float* __restrict__ pool,
                                  const float* __restrict__ gcnt,
                                  float* __restrict__ out) {
    int t = blockIdx.x * blockDim.x + threadIdx.x;
    if (t < NGRAPH * 64)
        out[t] = pool[t] / fmaxf(gcnt[t >> 6], 1.0f);
}

// ---------------------------------------------------------------------------
extern "C" void kernel_launch(void* const* d_in, const int* in_sizes, int n_in,
                              void* d_out, int out_size, void* d_ws, size_t ws_size,
                              hipStream_t stream) {
    const float* x     = (const float*)d_in[0];
    const int*   eidx  = (const int*)  d_in[1];
    const int*   batch = (const int*)  d_in[2];
    const float* Wl0 = (const float*)d_in[3];
    const float* Wr0 = (const float*)d_in[4];
    const float* b0  = (const float*)d_in[5];
    const float* Wl1 = (const float*)d_in[6];
    const float* Wr1 = (const float*)d_in[7];
    const float* b1  = (const float*)d_in[8];
    const float* Wl2 = (const float*)d_in[9];
    const float* Wr2 = (const float*)d_in[10];
    const float* b2  = (const float*)d_in[11];

    const int* src = eidx;
    const int* dst = eidx + NEDGES;

    // ---- workspace carve-out ----------------------------------------------
    char* ws = (char*)d_ws;
    size_t off = 0;
    auto bump = [&](size_t bytes) {
        char* p = ws + off;
        off += (bytes + 255) & ~(size_t)255;
        return p;
    };
    float*  cnt  = (float*)bump((size_t)NNODES * 4);
    float*  gcnt = (float*)bump((size_t)NGRAPH * 4);
    float*  pool = (float*)bump((size_t)NGRAPH * 64 * 4);
    __bf16* Pl0  = (__bf16*)bump((size_t)8 * 4 * 512 * 2);
    __bf16* Pr0  = (__bf16*)bump((size_t)8 * 4 * 512 * 2);
    __bf16* Pl1  = (__bf16*)bump((size_t)8 * 4 * 512 * 2);
    __bf16* Pr1  = (__bf16*)bump((size_t)8 * 4 * 512 * 2);
    __bf16* Pl2  = (__bf16*)bump((size_t)4 * 4 * 512 * 2);
    __bf16* Pr2  = (__bf16*)bump((size_t)4 * 4 * 512 * 2);
    float*  aggr = (float*)bump((size_t)NNODES * 128 * 4);
    float*  h1   = (float*)bump((size_t)NNODES * 128 * 4);
    float*  h2   = (float*)bump((size_t)NNODES * 128 * 4);
    float*  h3   = (float*)bump((size_t)NNODES * 64 * 4);
    (void)ws_size; (void)n_in; (void)in_sizes; (void)out_size;

    // ---- weight pre-pack (bf16 WMMA B-fragments) --------------------------
    pack_weights_kernel<<<4, 256, 0, stream>>>(Wl0, Pl0, 8, 4, 128);
    pack_weights_kernel<<<4, 256, 0, stream>>>(Wr0, Pr0, 8, 4, 128);
    pack_weights_kernel<<<4, 256, 0, stream>>>(Wl1, Pl1, 8, 4, 128);
    pack_weights_kernel<<<4, 256, 0, stream>>>(Wr1, Pr1, 8, 4, 128);
    pack_weights_kernel<<<2, 256, 0, stream>>>(Wl2, Pl2, 4, 4, 128);
    pack_weights_kernel<<<2, 256, 0, stream>>>(Wr2, Pr2, 4, 4, 128);

    // ---- degrees (shared by all layers) -----------------------------------
    hipMemsetAsync(cnt, 0, (size_t)NNODES * 4, stream);
    degree_kernel<<<1024, 256, 0, stream>>>(dst, cnt, NEDGES);

    const int scatterGrid = 8192;          // grid-stride over E*32 lanes
    const int gemmGrid    = NNODES / 16;   // 3125, exact

    // ---- layer 0: x -> h1 (ReLU) ------------------------------------------
    hipMemsetAsync(aggr, 0, (size_t)NNODES * 128 * 4, stream);
    scatter_add_kernel<<<scatterGrid, 256, 0, stream>>>(x, src, dst, aggr, NEDGES);
    sage_gemm_norm_kernel<8, true><<<gemmGrid, 32, 0, stream>>>(
        x, aggr, cnt, Pl0, Pr0, b0, h1);

    // ---- layer 1: h1 -> h2 (ReLU) -----------------------------------------
    hipMemsetAsync(aggr, 0, (size_t)NNODES * 128 * 4, stream);
    scatter_add_kernel<<<scatterGrid, 256, 0, stream>>>(h1, src, dst, aggr, NEDGES);
    sage_gemm_norm_kernel<8, true><<<gemmGrid, 32, 0, stream>>>(
        h1, aggr, cnt, Pl1, Pr1, b1, h2);

    // ---- layer 2: h2 -> h3 (no ReLU, dout=64) -----------------------------
    hipMemsetAsync(aggr, 0, (size_t)NNODES * 128 * 4, stream);
    scatter_add_kernel<<<scatterGrid, 256, 0, stream>>>(h2, src, dst, aggr, NEDGES);
    sage_gemm_norm_kernel<4, false><<<gemmGrid, 32, 0, stream>>>(
        h2, aggr, cnt, Pl2, Pr2, b2, h3);

    // ---- global mean pool -> d_out ----------------------------------------
    hipMemsetAsync(gcnt, 0, (size_t)NGRAPH * 4, stream);
    hipMemsetAsync(pool, 0, (size_t)NGRAPH * 64 * 4, stream);
    pool_accum_kernel<<<2048, 256, 0, stream>>>(h3, batch, pool, gcnt, NNODES);
    pool_final_kernel<<<(NGRAPH * 64 + 255) / 256, 256, 0, stream>>>(
        pool, gcnt, (float*)d_out);
}
